// QuantumCritic_81716047773722
// MI455X (gfx1250) — compile-verified, matrix operand order checked by reading, not run on hardware
//
#include <hip/hip_runtime.h>

typedef __attribute__((ext_vector_type(16))) _Float16 v16h;
typedef __attribute__((ext_vector_type(8)))  float    v8f;

union F16Frag { v16h v; uint4 q[2]; };

#define OBS 96
#define ACT 32
#define DIN 128
#define HID 256
#define HPAD 260

// ---------------- quantum helpers ----------------
// 4 batch elements per wave, interleaved. 8 complex amps per lane per element,
// amp index = (lane<<3)|j. bit P: P<3 -> in-lane bit of j ; P>=3 -> lane bit (P-3).

template<int P>
__device__ inline void apply_rx1(float sr[8], float si[8], int lane, float c, float s) {
  if constexpr (P < 3) {
    constexpr int m = 1 << P;
    #pragma unroll
    for (int j = 0; j < 8; ++j) {
      if (j & m) continue;
      const int j1 = j | m;
      float ar = sr[j], ai = si[j], br = sr[j1], bi = si[j1];
      sr[j]  = c*ar + s*bi;  si[j]  = c*ai - s*br;
      sr[j1] = c*br + s*ai;  si[j1] = c*bi - s*ar;
    }
  } else {
    constexpr int mask = 1 << (P - 3);
    #pragma unroll
    for (int j = 0; j < 8; ++j) {
      float pr = __shfl_xor(sr[j], mask, 32);
      float pi = __shfl_xor(si[j], mask, 32);
      float nr = c*sr[j] + s*pi;
      float ni = c*si[j] - s*pr;
      sr[j] = nr; si[j] = ni;
    }
  }
}

// encoding RX on all 4 interleaved states (per-element angles)
template<int P>
__device__ inline void enc_rx4(float sr[4][8], float si[4][8], int lane,
                               const float* encs, int rbase, int qi) {
  #pragma unroll
  for (int e = 0; e < 4; ++e) {
    float t = encs[(rbase + e) * 8 + qi];
    float s, c;
    __sincosf(0.5f * t, &s, &c);
    apply_rx1<P>(sr[e], si[e], lane, c, s);
  }
}

// fused U = RZ(tz)*RY(ty)*RX(tx) applied to all 4 states
template<int P>
__device__ inline void apply_u4(float sr[4][8], float si[4][8], int lane,
                                float tx, float ty, float tz) {
  float sx, cx, sy, cy, sz, cz;
  __sincosf(0.5f * tx, &sx, &cx);
  __sincosf(0.5f * ty, &sy, &cy);
  __sincosf(0.5f * tz, &sz, &cz);
  // M = RY * RX
  const float m00r =  cy * cx, m00i =  sy * sx;
  const float m01r = -sy * cx, m01i = -cy * sx;
  const float m10r =  sy * cx, m10i = -cy * sx;
  const float m11r =  cy * cx, m11i = -sy * sx;
  // U = RZ * M : row0 *= (cz - i sz), row1 *= (cz + i sz)
  const float u00r = cz*m00r + sz*m00i, u00i = cz*m00i - sz*m00r;
  const float u01r = cz*m01r + sz*m01i, u01i = cz*m01i - sz*m01r;
  const float u10r = cz*m10r - sz*m10i, u10i = cz*m10i + sz*m10r;
  const float u11r = cz*m11r - sz*m11i, u11i = cz*m11i + sz*m11r;

  if constexpr (P < 3) {
    constexpr int m = 1 << P;
    #pragma unroll
    for (int e = 0; e < 4; ++e) {
      #pragma unroll
      for (int j = 0; j < 8; ++j) {
        if (j & m) continue;
        const int j1 = j | m;
        float ar = sr[e][j],  ai = si[e][j];
        float br = sr[e][j1], bi = si[e][j1];
        sr[e][j]  = u00r*ar - u00i*ai + u01r*br - u01i*bi;
        si[e][j]  = u00r*ai + u00i*ar + u01r*bi + u01i*br;
        sr[e][j1] = u10r*ar - u10i*ai + u11r*br - u11i*bi;
        si[e][j1] = u10r*ai + u10i*ar + u11r*bi + u11i*br;
      }
    }
  } else {
    constexpr int mask = 1 << (P - 3);
    const int bit = (lane >> (P - 3)) & 1;
    // bit==0: new0 = u00*own + u01*part ; bit==1: new1 = u11*own + u10*part
    const float car = bit ? u11r : u00r, cai = bit ? u11i : u00i;
    const float cbr = bit ? u10r : u01r, cbi = bit ? u10i : u01i;
    #pragma unroll
    for (int e = 0; e < 4; ++e) {
      #pragma unroll
      for (int j = 0; j < 8; ++j) {
        float pr = __shfl_xor(sr[e][j], mask, 32);
        float pi = __shfl_xor(si[e][j], mask, 32);
        float orr = sr[e][j], oi = si[e][j];
        sr[e][j] = car*orr - cai*oi + cbr*pr - cbi*pi;
        si[e][j] = car*oi + cai*orr + cbr*pi + cbi*pr;
      }
    }
  }
}

// Composed CNOT block for gates (0,1),(1,2),(2,3),(3,4): pure lane-bit linear
// permutation. Source lane for the gathered state is lane ^ (lane>>1).
__device__ inline void cnot_lane_block4(float sr[4][8], float si[4][8], int lane) {
  const int srcl = (lane ^ (lane >> 1)) & 31;
  #pragma unroll
  for (int e = 0; e < 4; ++e) {
    #pragma unroll
    for (int j = 0; j < 8; ++j) {
      sr[e][j] = __shfl(sr[e][j], srcl, 32);
      si[e][j] = __shfl(si[e][j], srcl, 32);
    }
  }
}

template<int PC, int PT>
__device__ inline void apply_cnot4(float sr[4][8], float si[4][8], int lane) {
  if constexpr (PT < 3) {
    constexpr int m = 1 << PT;
    #pragma unroll
    for (int e = 0; e < 4; ++e) {
      #pragma unroll
      for (int j = 0; j < 8; ++j) {
        if (j & m) continue;
        const int j1 = j | m;
        bool ctrl;
        if constexpr (PC < 3) ctrl = ((j >> PC) & 1) != 0;
        else                  ctrl = ((lane >> (PC - 3)) & 1) != 0;
        if (ctrl) {
          float tr = sr[e][j]; sr[e][j] = sr[e][j1]; sr[e][j1] = tr;
          float ti = si[e][j]; si[e][j] = si[e][j1]; si[e][j1] = ti;
        }
      }
    }
  } else {
    constexpr int mask = 1 << (PT - 3);
    #pragma unroll
    for (int e = 0; e < 4; ++e) {
      #pragma unroll
      for (int j = 0; j < 8; ++j) {
        float pr = __shfl_xor(sr[e][j], mask, 32);
        float pi = __shfl_xor(si[e][j], mask, 32);
        bool ctrl;
        if constexpr (PC < 3) ctrl = ((j >> PC) & 1) != 0;
        else                  ctrl = ((lane >> (PC - 3)) & 1) != 0;
        if (ctrl) { sr[e][j] = pr; si[e][j] = pi; }
      }
    }
  }
}

// ---------------- prep: w1 (f32, [128][256]) -> f16 transposed [256][128] in ws ----------------
__global__ void qc_prep_w1(const float* __restrict__ w1a, const float* __restrict__ w1b,
                           unsigned short* __restrict__ w1t) {
  int tid = blockIdx.x * blockDim.x + threadIdx.x;     // 0..65535
  int critic = tid >> 15;
  int r = tid & 32767;
  int k = r >> 8;          // 0..127
  int n = r & 255;         // 0..255
  const float* w1 = critic ? w1b : w1a;
  _Float16 h = (_Float16)w1[k * 256 + n];
  w1t[critic * 32768 + n * 128 + k] = __builtin_bit_cast(unsigned short, h);
}

// ---------------- fused kernel: GEMM1(WMMA) -> enc -> quantum sim -> head ----------------
__global__ __launch_bounds__(256) void qc_main(
    const float* __restrict__ state, const float* __restrict__ action,
    const unsigned short* __restrict__ w1t_u,
    const float* __restrict__ b1a, const float* __restrict__ w2a, const float* __restrict__ b2a,
    const float* __restrict__ qrxa, const float* __restrict__ qrya, const float* __restrict__ qrza,
    const float* __restrict__ w3a, const float* __restrict__ b3a,
    const float* __restrict__ w4a, const float* __restrict__ b4a,
    const float* __restrict__ b1b, const float* __restrict__ w2b, const float* __restrict__ b2b,
    const float* __restrict__ qrxb, const float* __restrict__ qryb, const float* __restrict__ qrzb,
    const float* __restrict__ w3b, const float* __restrict__ b3b,
    const float* __restrict__ w4b, const float* __restrict__ b4b,
    float* __restrict__ out, int B) {
  __shared__ alignas(16) _Float16 xt[32][DIN];   // 8 KB
  __shared__ float hbuf[32][HPAD];               // ~33 KB (padded vs bank conflicts)
  __shared__ float w2s[HID * 8];                 // 8 KB
  __shared__ float b1s[HID];
  __shared__ float b2s[8];
  __shared__ float encs[32 * 8];
  __shared__ float w3s[8 * HID];                 // 8 KB
  __shared__ float b3s[HID];
  __shared__ float w4s[HID];

  const int t = threadIdx.x;
  const int critic = blockIdx.y;
  const int row0 = blockIdx.x * 32;

  const float* b1  = critic ? b1b  : b1a;
  const float* w2  = critic ? w2b  : w2a;
  const float* b2  = critic ? b2b  : b2a;
  const float* qx  = critic ? qrxb : qrxa;
  const float* qy  = critic ? qryb : qrya;
  const float* qz  = critic ? qrzb : qrza;
  const float* w3  = critic ? w3b  : w3a;
  const float* b3  = critic ? b3b  : b3a;
  const float* w4  = critic ? w4b  : w4a;
  const float  b4v = critic ? b4b[0] : b4a[0];
  const _Float16* w1t = reinterpret_cast<const _Float16*>(w1t_u) + critic * 32768;

  // ---- stage 0: cooperative loads ----
  #pragma unroll
  for (int it = 0; it < 16; ++it) {
    int idx = it * 256 + t;
    int r = idx >> 7, c = idx & 127;
    float v = (c < OBS) ? state[(row0 + r) * OBS + c]
                        : action[(row0 + r) * ACT + (c - OBS)];
    xt[r][c] = (_Float16)v;
  }
  #pragma unroll
  for (int it = 0; it < 8; ++it) {
    int idx = it * 256 + t;
    w2s[idx] = w2[idx];
    w3s[idx] = w3[idx];
  }
  b1s[t] = b1[t];
  b3s[t] = b3[t];
  w4s[t] = w4[t];
  if (t < 8) b2s[t] = b2[t];
  __syncthreads();

  // ---- stage 1: h = relu(x@w1 + b1) via v_wmma_f32_16x16x32_f16 ----
  const int lane = t & 31, wv = t >> 5;
  const int rg = wv & 1;        // 2 row groups of 16
  const int cg = wv >> 1;       // 4 col groups of 64
  const int lr = lane & 15, g = lane >> 4;

  v8f acc[4] = {};
  #pragma unroll
  for (int ks = 0; ks < 4; ++ks) {          // K = 128 / 32
    F16Frag a;
    const _Float16* ap = &xt[rg * 16 + lr][ks * 32 + g * 8];
    a.q[0] = *(const uint4*)ap;
    a.q[1] = *(const uint4*)(ap + 16);
    #pragma unroll
    for (int tt = 0; tt < 4; ++tt) {
      int n = cg * 64 + tt * 16 + lr;
      const _Float16* bp = w1t + n * 128 + ks * 32 + g * 8;
      F16Frag bf;
      bf.q[0] = *(const uint4*)bp;
      bf.q[1] = *(const uint4*)(bp + 16);
      acc[tt] = __builtin_amdgcn_wmma_f32_16x16x32_f16(
          false, a.v, false, bf.v, (short)0, acc[tt], false, false);
    }
  }
  #pragma unroll
  for (int tt = 0; tt < 4; ++tt) {
    #pragma unroll
    for (int i = 0; i < 8; ++i) {
      int r = rg * 16 + g * 8 + i;
      int c = cg * 64 + tt * 16 + lr;
      float v = acc[tt][i] + b1s[c];
      hbuf[r][c] = v > 0.f ? v : 0.f;
    }
  }
  __syncthreads();

  // ---- stage 2: enc = h @ w2 + b2  (one (row,q) per thread) ----
  {
    int r = t >> 3, q = t & 7;
    float s = b2s[q];
    #pragma unroll 8
    for (int k = 0; k < HID; ++k) s += hbuf[r][k] * w2s[k * 8 + q];
    encs[r * 8 + q] = s;
  }
  __syncthreads();

  // ---- stage 3: quantum sim, 4 batch elements per wave INTERLEAVED ----
  const int rbase = wv * 4;
  float sr[4][8], si[4][8];
  #pragma unroll
  for (int e = 0; e < 4; ++e) {
    #pragma unroll
    for (int j = 0; j < 8; ++j) { sr[e][j] = 0.f; si[e][j] = 0.f; }
    if (lane == 0) sr[e][0] = 1.f;
  }

  // encoding RX: qubit i -> bit 7-i (per-element angles)
  enc_rx4<7>(sr, si, lane, encs, rbase, 0);
  enc_rx4<6>(sr, si, lane, encs, rbase, 1);
  enc_rx4<5>(sr, si, lane, encs, rbase, 2);
  enc_rx4<4>(sr, si, lane, encs, rbase, 3);
  enc_rx4<3>(sr, si, lane, encs, rbase, 4);
  enc_rx4<2>(sr, si, lane, encs, rbase, 5);
  enc_rx4<1>(sr, si, lane, encs, rbase, 6);
  enc_rx4<0>(sr, si, lane, encs, rbase, 7);

  for (int l = 0; l < 3; ++l) {
    const float *ax = qx + l * 8, *ay = qy + l * 8, *az = qz + l * 8;
    // fused U = RZ*RY*RX per qubit, wave-uniform, shared by all 4 elements
    apply_u4<7>(sr, si, lane, ax[0], ay[0], az[0]);
    apply_u4<6>(sr, si, lane, ax[1], ay[1], az[1]);
    apply_u4<5>(sr, si, lane, ax[2], ay[2], az[2]);
    apply_u4<4>(sr, si, lane, ax[3], ay[3], az[3]);
    apply_u4<3>(sr, si, lane, ax[4], ay[4], az[4]);
    apply_u4<2>(sr, si, lane, ax[5], ay[5], az[5]);
    apply_u4<1>(sr, si, lane, ax[6], ay[6], az[6]);
    apply_u4<0>(sr, si, lane, ax[7], ay[7], az[7]);
    // CNOT ring: (0,1)(1,2)(2,3)(3,4) composed into one lane gather,
    // then (4,5)=<3,2>, (5,6)=<2,1>, (6,7)=<1,0>, (7,0)=<0,7> sequentially.
    cnot_lane_block4(sr, si, lane);
    apply_cnot4<3, 2>(sr, si, lane);
    apply_cnot4<2, 1>(sr, si, lane);
    apply_cnot4<1, 0>(sr, si, lane);
    apply_cnot4<0, 7>(sr, si, lane);
  }

  // ---- stage 4: expectations + head, per element ----
  #pragma unroll
  for (int e = 0; e < 4; ++e) {
    float prob[8], ptot = 0.f;
    #pragma unroll
    for (int j = 0; j < 8; ++j) {
      prob[j] = sr[e][j] * sr[e][j] + si[e][j] * si[e][j];
      ptot += prob[j];
    }
    float ex[8];
    #pragma unroll
    for (int k = 0; k < 8; ++k) {
      const int p = 7 - k;
      if (p >= 3) {
        ex[k] = ((lane >> (p - 3)) & 1) ? -ptot : ptot;
      } else {
        float s = 0.f;
        #pragma unroll
        for (int j = 0; j < 8; ++j) s += ((j >> p) & 1) ? -prob[j] : prob[j];
        ex[k] = s;
      }
    }
    #pragma unroll
    for (int m = 1; m < 32; m <<= 1) {
      #pragma unroll
      for (int k = 0; k < 8; ++k) ex[k] += __shfl_xor(ex[k], m, 32);
    }

    float qp = 0.f;
    #pragma unroll
    for (int nn = 0; nn < 8; ++nn) {
      int n = nn * 32 + lane;
      float hv = b3s[n];
      #pragma unroll
      for (int k = 0; k < 8; ++k) hv += ex[k] * w3s[k * HID + n];
      hv = hv > 0.f ? hv : 0.f;
      qp += hv * w4s[n];
    }
    #pragma unroll
    for (int m = 1; m < 32; m <<= 1) qp += __shfl_xor(qp, m, 32);
    if (lane == 0) out[critic * B + (row0 + rbase + e)] = qp + b4v;
  }
}

extern "C" void kernel_launch(void* const* d_in, const int* in_sizes, int n_in,
                              void* d_out, int out_size, void* d_ws, size_t ws_size,
                              hipStream_t stream) {
  const float* state  = (const float*)d_in[0];
  const float* action = (const float*)d_in[1];
  const float* w1a = (const float*)d_in[2];
  const float* b1a = (const float*)d_in[3];
  const float* w2a = (const float*)d_in[4];
  const float* b2a = (const float*)d_in[5];
  const float* qrxa = (const float*)d_in[6];
  const float* qrya = (const float*)d_in[7];
  const float* qrza = (const float*)d_in[8];
  const float* w3a = (const float*)d_in[9];
  const float* b3a = (const float*)d_in[10];
  const float* w4a = (const float*)d_in[11];
  const float* b4a = (const float*)d_in[12];
  const float* w1b = (const float*)d_in[13];
  const float* b1b = (const float*)d_in[14];
  const float* w2b = (const float*)d_in[15];
  const float* b2b = (const float*)d_in[16];
  const float* qrxb = (const float*)d_in[17];
  const float* qryb = (const float*)d_in[18];
  const float* qrzb = (const float*)d_in[19];
  const float* w3b = (const float*)d_in[20];
  const float* b3b = (const float*)d_in[21];
  const float* w4b = (const float*)d_in[22];
  const float* b4b = (const float*)d_in[23];

  const int B = in_sizes[0] / OBS;                 // 65536
  unsigned short* w1t = (unsigned short*)d_ws;     // 2 * 256*128 f16 = 128 KB

  qc_prep_w1<<<256, 256, 0, stream>>>(w1a, w1b, w1t);

  dim3 grid(B / 32, 2);
  qc_main<<<grid, 256, 0, stream>>>(
      state, action, w1t,
      b1a, w2a, b2a, qrxa, qrya, qrza, w3a, b3a, w4a, b4a,
      b1b, w2b, b2b, qrxb, qryb, qrzb, w3b, b3b, w4b, b4b,
      (float*)d_out, B);
}
